// PVQVAE_66108136620721
// MI455X (gfx1250) — compile-verified, matrix operand order checked by reading, not run on hardware
//
// PVQVAE fused forward for MI455X (gfx1250, CDNA5, wave32).
// One workgroup (256 threads = 8 wave32) per batch element; entire per-sample
// state lives in LDS; all GEMMs via v_wmma_f32_16x16x32_bf16 (fp32 accum).
// Fragments are loaded as 2x b128 per lane (contiguous runs of the documented
// 16-bit WMMA layout); weights are pre-transposed to bf16 in d_ws so B
// fragments are also 2x b128; V is kept transposed in LDS for the same reason.
#include <hip/hip_runtime.h>
#include <math.h>

#define F_TOK   128
#define D_MODEL 256
#define N_HEAD  8
#define D_HEAD  32
#define L_LAT   32
#define K_CODE  1024
#define D_HID   1024
#define N_DEPTH 4
#define NWAVE   8
#define EPSLN   1e-5f
#define BETA_C  0.25f

typedef __bf16 bf16;
typedef __attribute__((ext_vector_type(16))) __bf16 v16bf;
typedef __attribute__((ext_vector_type(8)))  __bf16 v8bf;   // 16 bytes
typedef __attribute__((ext_vector_type(8)))  float  v8f;

// ---------------- LDS pool layout (bf16 elements; 1KB = 512 elems) ----------
#define EKB 512
#define OFF_X   0                 // residual stream: 32x256 (enc/self) or 128x256 (dec)
// encoder phase (peak usage: 288KB)
#define EOFF_KV (16*EKB)          // fe 128x256
#define EOFF_K  (80*EKB)          // K 128x256
#define EOFF_V  (144*EKB)         // Vt 256x128
#define EOFF_Q  (208*EKB)         // Q/O 32x256
#define EOFF_T  (224*EKB)         // scores 8x32x128 / mlp-hidden 32x1024
// self-attn phase
#define SOFF_K  (16*EKB)
#define SOFF_V  (32*EKB)          // Vt 256x128 (64KB)
#define SOFF_Q  (96*EKB)
#define SOFF_T  (112*EKB)
// decoder phase
#define DOFF_KV (64*EKB)          // qz 32x256 (written by VQ phase)
#define DOFF_K  (80*EKB)
#define DOFF_V  (96*EKB)          // Vt (64KB)
#define DOFF_Q  (160*EKB)
#define DOFF_T  (176*EKB)
#define BYTE_SC  (128*1024)       // fp32 VQ score chunk 32x256
#define BYTE_RED (288*1024)       // small reduction scratch
#define SMEM_BYTES (288*1024 + 1024)

// ------------------------- WMMA fragment helpers ----------------------------
// 16-bit A (16x32 MxK) layout (ISA 7.12.2): lane<16 -> row=lane, K=koff..koff+7
// and koff+16..koff+23 with koff=0; lane>=16 -> row=lane-16, koff=8.
// Each lane's 16 elements are two contiguous 8-element (16B) runs -> 2x b128.
// The same pattern loads B when the source holds B^T row-major
// (lane -> column of B -> row of source).
__device__ __forceinline__ v16bf load_frag_rm(const bf16* src, int ld, int lane) {
  const bf16* p = src + (lane & 15) * ld + ((lane & 16) ? 8 : 0);
  const v8bf lo = *(const v8bf*)(p);       // K = koff .. koff+7
  const v8bf hi = *(const v8bf*)(p + 16);  // K = koff+16 .. koff+23
  union { v16bf v; v8bf h[2]; } u;
  u.h[0] = lo;
  u.h[1] = hi;
  return u.v;
}

// C/D 16x16 f32: VGPR r holds M=r (lanes 0-15, N=lane) and M=r+8 (lanes 16-31).
__device__ __forceinline__ void cstore_bf16(bf16* dst, int ld, const v8f& c, int lane) {
  const int col = lane & 15;
  const int rof = (lane & 16) ? 8 : 0;
#pragma unroll
  for (int r = 0; r < 8; ++r) dst[(r + rof) * ld + col] = (bf16)c[r];
}

// Transposed C store: dstT[col][row]; each lane's 8 rows are contiguous -> 1x b128.
__device__ __forceinline__ void cstore_bf16_t(bf16* dstT, int ldT, int colbase, int rowbase,
                                              const v8f& c, int lane) {
  const int col  = colbase + (lane & 15);
  const int row0 = rowbase + ((lane & 16) ? 8 : 0);
  v8bf pk;
#pragma unroll
  for (int r = 0; r < 8; ++r) pk[r] = (bf16)c[r];
  *(v8bf*)(dstT + col * ldT + row0) = pk;
}

__device__ __forceinline__ v8f wmma_bf16(const v16bf& a, const v16bf& b, const v8f& c) {
  return __builtin_amdgcn_wmma_f32_16x16x32_bf16(false, a, false, b, (short)0, c, false, false);
}

// Generic GEMM: A (MxKdim bf16 row-major, LDS) x B (KdimxN) with B supplied as
// B^T row-major (NxKdim, global bf16 weights) -> ep(tile).
template <typename Ep>
__device__ __forceinline__ void gemm_bf16(const bf16* A, int lda, const bf16* Bt, int ldbt,
                                          int M, int N, int Kdim, int wave, int lane, Ep ep) {
  const int ntn = N >> 4;
  const int ntiles = (M >> 4) * ntn;
  for (int t = wave; t < ntiles; t += NWAVE) {
    const int tm = t / ntn, tn = t - tm * ntn;
    v8f acc = {0.f, 0.f, 0.f, 0.f, 0.f, 0.f, 0.f, 0.f};
    const bf16* ap = A + tm * 16 * lda;
    const bf16* bp = Bt + tn * 16 * ldbt;
    for (int k = 0; k < Kdim; k += 32) {
      __builtin_prefetch(bp + k + 32, 0, 0);  // next weight k-panel -> global_prefetch
      v16bf af = load_frag_rm(ap + k, lda, lane);
      v16bf bf = load_frag_rm(bp + k, ldbt, lane);
      acc = wmma_bf16(af, bf, acc);
    }
    ep(tm, tn, acc);
  }
}

// ------------------------- reductions / normalizations ----------------------
__device__ __forceinline__ float wave_sum(float v) {
#pragma unroll
  for (int o = 16; o; o >>= 1) v += __shfl_xor(v, o, 32);
  return v;
}
__device__ __forceinline__ float wave_max(float v) {
#pragma unroll
  for (int o = 16; o; o >>= 1) v = fmaxf(v, __shfl_xor(v, o, 32));
  return v;
}

__device__ void ln_rows(bf16* X, int rows, int W, int ld, const float* g, const float* bta,
                        float scale, int wave, int lane) {
  for (int r = wave; r < rows; r += NWAVE) {
    bf16* row = X + r * ld;
    float s = 0.f;
    for (int c = lane; c < W; c += 32) s += (float)row[c];
    const float mean = wave_sum(s) / (float)W;
    float v = 0.f;
    for (int c = lane; c < W; c += 32) { float d = (float)row[c] - mean; v += d * d; }
    const float rstd = rsqrtf(wave_sum(v) / (float)W + EPSLN);
    for (int c = lane; c < W; c += 32) {
      float val = ((float)row[c] - mean) * rstd * g[c] + bta[c];
      row[c] = (bf16)(val * scale);
    }
  }
}

// Per-head LN: contiguous 32-wide segments (head dim), one segment per wave pass.
__device__ void ln_heads(bf16* X, int rows, int ld, const float* g, const float* bta,
                         float scale, int wave, int lane) {
  const int nseg = rows * N_HEAD;
  for (int sg = wave; sg < nseg; sg += NWAVE) {
    const int r = sg >> 3, h = sg & 7;
    bf16* p = X + r * ld + h * D_HEAD;
    const float x = (float)p[lane];
    const float mean = wave_sum(x) * (1.f / 32.f);
    const float d = x - mean;
    const float rstd = rsqrtf(wave_sum(d * d) * (1.f / 32.f) + EPSLN);
    p[lane] = (bf16)((d * rstd * g[lane] + bta[lane]) * scale);
  }
}

// ------------------------------- layer params -------------------------------
struct LayerW {
  const bf16 *Wq, *Wk, *Wv, *Wo, *W1, *W2;                 // bf16 W^T (workspace)
  const float *qn_g, *qn_b, *kn_g, *kn_b, *n_g, *n_b;      // fp32 (inputs)
  const float *b1, *ln_g, *ln_b, *b2;
};

struct Params {
  const float *x, *tok, *pos, *lq, *dq, *proj;
  const bf16* cb;     // codebook row-major (acts as B^T)
  const float* cbn;   // codebook row sqnorms
  LayerW enc, dec, blk[N_DEPTH];
  float* recon;
  float* loss;
};

// --------------------------- one transformer block --------------------------
// X: Mx256 residual stream (bf16 LDS). KV: Nx256 (may alias X for self-attn).
// Vt: transposed value cache 256x128 (ld=128).
__device__ void transformer_block(bf16* X, int M, const bf16* KV, int N,
                                  bf16* Kb, bf16* Vt, bf16* Qb, bf16* Tb,
                                  const LayerW& w, int wave, int lane) {
  // K,V projections over all N kv tokens (V stored transposed: Vt[d][f])
  gemm_bf16(KV, D_MODEL, w.Wk, D_MODEL, N, D_MODEL, D_MODEL, wave, lane,
            [&](int tm, int tn, v8f acc) { cstore_bf16(Kb + tm * 16 * D_MODEL + tn * 16, D_MODEL, acc, lane); });
  gemm_bf16(KV, D_MODEL, w.Wv, D_MODEL, N, D_MODEL, D_MODEL, wave, lane,
            [&](int tm, int tn, v8f acc) { cstore_bf16_t(Vt, F_TOK, tn * 16, tm * 16, acc, lane); });
  __syncthreads();
  ln_heads(Kb, N, D_MODEL, w.kn_g, w.kn_b, 1.f, wave, lane);
  __syncthreads();

  for (int m0 = 0; m0 < M; m0 += 32) {
    bf16* Xt = X + m0 * D_MODEL;
    // Q tile (32x256)
    gemm_bf16(Xt, D_MODEL, w.Wq, D_MODEL, 32, D_MODEL, D_MODEL, wave, lane,
              [&](int tm, int tn, v8f acc) { cstore_bf16(Qb + tm * 16 * D_MODEL + tn * 16, D_MODEL, acc, lane); });
    __syncthreads();
    ln_heads(Qb, 32, D_MODEL, w.qn_g, w.qn_b, 0.17677669529663687f /*HD^-0.5 folded*/, wave, lane);
    __syncthreads();

    // scores[h] = q_h (32x32) @ k_h^T (32xN), bf16 into Tb (stride 128)
    {
      const int perh = 2 * (N >> 4);
      const int tiles = N_HEAD * perh;
      for (int t = wave; t < tiles; t += NWAVE) {
        const int h = t / perh, rem = t - h * perh;
        const int tm = rem / (N >> 4), tn = rem - tm * (N >> 4);
        v16bf a = load_frag_rm(Qb + tm * 16 * D_MODEL + h * D_HEAD, D_MODEL, lane);
        v16bf b = load_frag_rm(Kb + tn * 16 * D_MODEL + h * D_HEAD, D_MODEL, lane); // B = k^T
        v8f acc = {0.f, 0.f, 0.f, 0.f, 0.f, 0.f, 0.f, 0.f};
        acc = wmma_bf16(a, b, acc);
        cstore_bf16(Tb + (h * 32 + tm * 16) * 128 + tn * 16, 128, acc, lane);
      }
    }
    __syncthreads();
    // softmax over kv dim (rows: 8 heads * 32 q)
    for (int rr = wave; rr < N_HEAD * 32; rr += NWAVE) {
      bf16* row = Tb + rr * 128;
      float m = -3.4e38f;
      for (int c = lane; c < N; c += 32) m = fmaxf(m, (float)row[c]);
      m = wave_max(m);
      float s = 0.f;
      for (int c = lane; c < N; c += 32) s += __expf((float)row[c] - m);
      const float inv = 1.f / wave_sum(s);
      for (int c = lane; c < N; c += 32) row[c] = (bf16)(__expf((float)row[c] - m) * inv);
    }
    __syncthreads();
    // o[h] = attn_h (32xN) @ v_h (Nx32); B^T rows live in Vt -> 2x b128 loads
    {
      const int tiles = N_HEAD * 2 * (D_HEAD >> 4); // 8*2*2
      for (int t = wave; t < tiles; t += NWAVE) {
        const int h = t >> 2, tm = (t >> 1) & 1, tn = t & 1;
        v8f acc = {0.f, 0.f, 0.f, 0.f, 0.f, 0.f, 0.f, 0.f};
        for (int k = 0; k < N; k += 32) {
          v16bf a = load_frag_rm(Tb + (h * 32 + tm * 16) * 128 + k, 128, lane);
          v16bf b = load_frag_rm(Vt + (h * D_HEAD + tn * 16) * F_TOK + k, F_TOK, lane);
          acc = wmma_bf16(a, b, acc);
        }
        cstore_bf16(Qb + tm * 16 * D_MODEL + h * D_HEAD + tn * 16, D_MODEL, acc, lane); // O
      }
    }
    __syncthreads();
    ln_rows(Qb, 32, D_MODEL, D_MODEL, w.n_g, w.n_b, 1.f, wave, lane);
    __syncthreads();
    // x += ln(o) @ Wo
    gemm_bf16(Qb, D_MODEL, w.Wo, D_MODEL, 32, D_MODEL, D_MODEL, wave, lane,
              [&](int tm, int tn, v8f acc) {
                const int col = tn * 16 + (lane & 15);
                const int rof = (lane & 16) ? 8 : 0;
#pragma unroll
                for (int r = 0; r < 8; ++r) {
                  const int row = tm * 16 + r + rof;
                  Xt[row * D_MODEL + col] = (bf16)(acc[r] + (float)Xt[row * D_MODEL + col]);
                }
              });
    __syncthreads();
  }

  // MLP: x += W2( ln( gelu(x@W1 + b1) ) ) + b2
  for (int m0 = 0; m0 < M; m0 += 32) {
    bf16* Xt = X + m0 * D_MODEL;
    gemm_bf16(Xt, D_MODEL, w.W1, D_MODEL, 32, D_HID, D_MODEL, wave, lane,
              [&](int tm, int tn, v8f acc) {
                const int col = tn * 16 + (lane & 15);
                const int rof = (lane & 16) ? 8 : 0;
#pragma unroll
                for (int r = 0; r < 8; ++r) {
                  float v = acc[r] + w.b1[col];
                  v = 0.5f * v * (1.f + erff(v * 0.7071067811865476f)); // exact gelu
                  Tb[(tm * 16 + r + rof) * D_HID + col] = (bf16)v;
                }
              });
    __syncthreads();
    ln_rows(Tb, 32, D_HID, D_HID, w.ln_g, w.ln_b, 1.f, wave, lane);
    __syncthreads();
    gemm_bf16(Tb, D_HID, w.W2, D_HID, 32, D_MODEL, D_HID, wave, lane,
              [&](int tm, int tn, v8f acc) {
                const int col = tn * 16 + (lane & 15);
                const int rof = (lane & 16) ? 8 : 0;
#pragma unroll
                for (int r = 0; r < 8; ++r) {
                  const int row = tm * 16 + r + rof;
                  Xt[row * D_MODEL + col] =
                      (bf16)(acc[r] + w.b2[col] + (float)Xt[row * D_MODEL + col]);
                }
              });
    __syncthreads();
  }
}

// ------------------------------- main kernel --------------------------------
__global__ __launch_bounds__(256) void pvqvae_fused(Params p) {
  extern __shared__ char smem_raw[];
  bf16* pool = (bf16*)smem_raw;
  const int b    = blockIdx.x;
  const int tid  = threadIdx.x;
  const int wave = tid >> 5;
  const int lane = tid & 31;

  bf16* sX = pool + OFF_X;

  // ---- encoder: fe = x[b,f]*tok + pos ; lat = block(lq, fe, fe) ----
  {
    bf16* sKV = pool + EOFF_KV;
    for (int i = tid; i < F_TOK * D_MODEL; i += 256) {
      const int f = i >> 8;
      sKV[i] = (bf16)(p.x[b * F_TOK + f] * p.tok[i] + p.pos[i]);
    }
    for (int i = tid; i < L_LAT * D_MODEL; i += 256) sX[i] = (bf16)p.lq[i];
    __syncthreads();
    transformer_block(sX, L_LAT, sKV, F_TOK,
                      pool + EOFF_K, pool + EOFF_V, pool + EOFF_Q, pool + EOFF_T,
                      p.enc, wave, lane);
  }

  // ---- self-attention stack ----
  for (int li = 0; li < N_DEPTH; ++li) {
    transformer_block(sX, L_LAT, sX, L_LAT,
                      pool + SOFF_K, pool + SOFF_V, pool + SOFF_Q, pool + SOFF_T,
                      p.blk[li], wave, lane);
  }

  // ---- vector quantization: argmin_c ||c||^2 - 2 z.c ----
  {
    float* sc    = (float*)(smem_raw + BYTE_SC);   // 32x256 score chunk
    float* bestv = (float*)(smem_raw + BYTE_RED);  // 32
    int*   besti = (int*)(bestv + 32);             // 32
    float* redb  = (float*)(besti + 32);           // NWAVE
    if (tid < L_LAT) { bestv[tid] = 3.4e38f; besti[tid] = 0; }
    __syncthreads();

    for (int c0 = 0; c0 < K_CODE; c0 += 256) {
      for (int t = wave; t < 32; t += NWAVE) {  // (32/16)x(256/16) tiles
        const int tm = t >> 4, tn = t & 15;
        v8f acc = {0.f, 0.f, 0.f, 0.f, 0.f, 0.f, 0.f, 0.f};
        for (int k = 0; k < D_MODEL; k += 32) {
          v16bf a = load_frag_rm(sX + tm * 16 * D_MODEL + k, D_MODEL, lane);
          v16bf bfr = load_frag_rm(p.cb + (c0 + tn * 16) * D_MODEL + k, D_MODEL, lane); // B = cb^T
          acc = wmma_bf16(a, bfr, acc);
        }
        const int col = tn * 16 + (lane & 15);
        const int rof = (lane & 16) ? 8 : 0;
        const float cn = p.cbn[c0 + col];
#pragma unroll
        for (int r = 0; r < 8; ++r) sc[(tm * 16 + r + rof) * 256 + col] = cn - 2.f * acc[r];
      }
      __syncthreads();
      for (int r = wave; r < L_LAT; r += NWAVE) {
        float bv = 3.4e38f; int bi = c0;
        for (int c = lane; c < 256; c += 32) {
          const float v = sc[r * 256 + c];
          if (v < bv) { bv = v; bi = c0 + c; }
        }
#pragma unroll
        for (int o = 16; o; o >>= 1) {
          const float ov = __shfl_xor(bv, o, 32);
          const int   oi = __shfl_xor(bi, o, 32);
          if (ov < bv || (ov == bv && oi < bi)) { bv = ov; bi = oi; }
        }
        if (lane == 0 && (bv < bestv[r] || (bv == bestv[r] && bi < besti[r]))) {
          bestv[r] = bv; besti[r] = bi;
        }
      }
      __syncthreads();
    }

    // qz gather (straight-through == qz in forward) + vq loss = (1+beta)*mse
    bf16* qz = pool + DOFF_KV;
    float part = 0.f;
    for (int i = tid; i < L_LAT * D_MODEL; i += 256) {
      const int r = i >> 8, d = i & 255;
      const bf16 q = p.cb[besti[r] * D_MODEL + d];
      qz[i] = q;
      const float df = (float)q - (float)sX[i];
      part += df * df;
    }
    part = wave_sum(part);
    if (lane == 0) redb[wave] = part;
    __syncthreads();
    if (tid == 0) {
      float s = 0.f;
      for (int w2 = 0; w2 < NWAVE; ++w2) s += redb[w2];
      p.loss[b] = s * (1.f + BETA_C) / (float)(L_LAT * D_MODEL);
    }
    __syncthreads();
  }

  // ---- decoder: out = block(dq, qz, qz) ----
  for (int i = tid; i < F_TOK * D_MODEL; i += 256) sX[i] = (bf16)p.dq[i];
  __syncthreads();
  transformer_block(sX, F_TOK, pool + DOFF_KV, L_LAT,
                    pool + DOFF_K, pool + DOFF_V, pool + DOFF_Q, pool + DOFF_T,
                    p.dec, wave, lane);

  // ---- recon[b,f] = sum_d out[f,d] * proj[f,d] ----
  for (int f = wave; f < F_TOK; f += NWAVE) {
    float s = 0.f;
    for (int d = lane; d < D_MODEL; d += 32)
      s += (float)sX[f * D_MODEL + d] * p.proj[f * D_MODEL + d];
    s = wave_sum(s);
    if (lane == 0) p.recon[b * F_TOK + f] = s;
  }
}

// ----------------------------- prologue kernels -----------------------------
__global__ void cvt_f32_bf16(const float* __restrict__ src, bf16* __restrict__ dst, int n) {
  for (int i = blockIdx.x * 256 + threadIdx.x; i < n; i += gridDim.x * 256)
    dst[i] = (bf16)src[i];
}

// dst (NxK bf16, row-major) = transpose of src (KxN fp32, row-major)
__global__ void cvt_t_f32_bf16(const float* __restrict__ src, bf16* __restrict__ dst,
                               int K, int N) {
  for (int i = blockIdx.x * 256 + threadIdx.x; i < K * N; i += gridDim.x * 256) {
    const int n = i / K, k = i - n * K;  // i indexes dst
    dst[i] = (bf16)src[k * N + n];
  }
}

__global__ void cb_norms(const float* __restrict__ cb, float* __restrict__ out) {
  __shared__ float sh[64];
  const int r = blockIdx.x;
  float s = 0.f;
  for (int d = threadIdx.x; d < D_MODEL; d += 64) { const float v = cb[r * D_MODEL + d]; s += v * v; }
  sh[threadIdx.x] = s;
  __syncthreads();
  for (int o = 32; o; o >>= 1) { if (threadIdx.x < o) sh[threadIdx.x] += sh[threadIdx.x + o]; __syncthreads(); }
  if (threadIdx.x == 0) out[r] = sh[0];
}

// ------------------------------- host launch --------------------------------
// Leaf order inside each param dict (jax tree: sorted keys):
// 0:W1 1:W2 2:Wk 3:Wo 4:Wq 5:Wv 6:b1 7:b2 8:kn_b 9:kn_g 10:ln_b 11:ln_g 12:n_b 13:n_g 14:qn_b 15:qn_g
// Top-level (insertion order): 0:x 1:tok 2:pos 3:lq 4:dq 5:proj 6:cb 7..22:enc 23..38:dec 39..54:blocks(x4)
#define LAYER_ELEMS (4 * 65536 + 2 * 262144)  // 786432 bf16 per layer in ws

static void fill_layer(LayerW& w, void* const* din, int base, int st, const bf16* wsl) {
  auto f = [&](int leaf, int elems) { return (const float*)din[base + leaf] + (size_t)st * elems; };
  w.qn_g = f(15, D_HEAD); w.qn_b = f(14, D_HEAD);
  w.kn_g = f(9, D_HEAD);  w.kn_b = f(8, D_HEAD);
  w.n_g  = f(13, D_MODEL); w.n_b = f(12, D_MODEL);
  w.b1   = f(6, D_HID);   w.b2   = f(7, D_MODEL);
  w.ln_g = f(11, D_HID);  w.ln_b = f(10, D_HID);
  w.Wq = wsl;              w.Wk = wsl + 65536;  w.Wv = wsl + 2 * 65536;
  w.Wo = wsl + 3 * 65536;  w.W1 = wsl + 262144; w.W2 = wsl + 524288;
}

extern "C" void kernel_launch(void* const* d_in, const int* in_sizes, int n_in,
                              void* d_out, int out_size, void* d_ws, size_t ws_size,
                              hipStream_t stream) {
  (void)n_in; (void)out_size; (void)ws_size;
  const int B = in_sizes[0] / F_TOK;

  bf16* wsb = (bf16*)d_ws;
  // ws layout: 6 layers (enc, blk0..3, dec) * 786432 bf16 (all W^T), then cb bf16, then cbn fp32
  bf16*  cbb = wsb + (size_t)6 * LAYER_ELEMS;
  float* cbn = (float*)(cbb + (size_t)K_CODE * D_MODEL);

  // --- convert+transpose weights fp32 -> bf16 W^T in ws ---
  const int   bigleaf[6] = {4, 2, 5, 3, 0, 1};                  // Wq Wk Wv Wo W1 W2
  const int   bigK[6]    = {256, 256, 256, 256, 256, 1024};     // in-dim
  const int   bigN[6]    = {256, 256, 256, 256, 1024, 256};     // out-dim
  const size_t bigoff[6] = {0, 65536, 131072, 196608, 262144, 524288};
  const int   lbase[6]   = {7, 39, 39, 39, 39, 23};             // enc, blk0..3, dec
  const int   lstk[6]    = {0, 0, 1, 2, 3, 0};
  for (int li = 0; li < 6; ++li)
    for (int wi = 0; wi < 6; ++wi) {
      const int n = bigK[wi] * bigN[wi];
      const float* src = (const float*)d_in[lbase[li] + bigleaf[wi]] + (size_t)lstk[li] * n;
      bf16* dst = wsb + (size_t)li * LAYER_ELEMS + bigoff[wi];
      cvt_t_f32_bf16<<<512, 256, 0, stream>>>(src, dst, bigK[wi], bigN[wi]);
    }
  cvt_f32_bf16<<<512, 256, 0, stream>>>((const float*)d_in[6], cbb, K_CODE * D_MODEL);
  cb_norms<<<K_CODE, 64, 0, stream>>>((const float*)d_in[6], cbn);

  // --- params ---
  Params p;
  p.x = (const float*)d_in[0]; p.tok = (const float*)d_in[1]; p.pos = (const float*)d_in[2];
  p.lq = (const float*)d_in[3]; p.dq = (const float*)d_in[4]; p.proj = (const float*)d_in[5];
  p.cb = cbb; p.cbn = cbn;
  fill_layer(p.enc, d_in, 7, 0, wsb);
  for (int i = 0; i < N_DEPTH; ++i) fill_layer(p.blk[i], d_in, 39, i, wsb + (size_t)(1 + i) * LAYER_ELEMS);
  fill_layer(p.dec, d_in, 23, 0, wsb + (size_t)5 * LAYER_ELEMS);
  p.recon = (float*)d_out;
  p.loss  = p.recon + (size_t)B * F_TOK;

  (void)hipFuncSetAttribute((const void*)pvqvae_fused,
                            hipFuncAttributeMaxDynamicSharedMemorySize, SMEM_BYTES);
  pvqvae_fused<<<B, 256, SMEM_BYTES, stream>>>(p);
}
